// RNN_49143015801230
// MI455X (gfx1250) — matrix operand order repaired and checked by reading, not compiled, not run on hardware
//
#include <hip/hip_runtime.h>
#include <math.h>
#include <stdint.h>

// ---------------------------------------------------------------------------
// GRU decoder for MI455X (gfx1250, wave32, WMMA).
//   B=64, T=64, E=256, F=256, H=512, V=10000, D=H+E+F=1024
// Strategy:
//   * bf16 WMMA (v_wmma_f32_16x16x32_bf16) for all GEMMs, fp32 accumulate.
//   * x-projection (time-invariant half of cat@W) hoisted into one big GEMM.
//   * per-step GEMM only state@W_state ([64,512]x[512,1536]).
//   * logits batched into one [4096,512]x[512,10000] GEMM with fused bias.
//   * NT template parameter: branch-free inner loop (EXEC all-1s around
//     every WMMA, no spills); ragged N handled by a separate tail launch.
//   * All GEMM memory ops forced to address-space(1) so they lower to
//     global_load/store (LOADcnt/STOREcnt only) instead of flat (which
//     also ticks DScnt and goes through LDS arbitration).
// ---------------------------------------------------------------------------

typedef __attribute__((ext_vector_type(8)))  __bf16 bf16x8;
typedef __attribute__((ext_vector_type(16))) __bf16 bf16x16;
typedef __attribute__((ext_vector_type(8)))  float  f32x8;

#define B_    64
#define T_    64
#define E_    256
#define F_    256
#define H_    512
#define V_    10000
#define G3_   1536          // 3 gates * H
#define BT_   4096          // B*T
#define KD_   512           // K of every GEMM (state half / x half / H)

// Global (addrspace 1) access helpers: generic->global bit patterns are
// identical on AMDGPU, so the integer round-trip cast is safe and forces
// global_load_b128 / global_store_b32 codegen.
__device__ __forceinline__ bf16x8 gload8(const __bf16* p) {
  return *(const __attribute__((address_space(1))) bf16x8*)(uintptr_t)p;
}
__device__ __forceinline__ float gloadf(const float* p) {
  return *(const __attribute__((address_space(1))) float*)(uintptr_t)p;
}
__device__ __forceinline__ void gstoref(float* p, float v) {
  *(__attribute__((address_space(1))) float*)(uintptr_t)p = v;
}

__device__ __forceinline__ bf16x16 combine16(bf16x8 lo, bf16x8 hi) {
  return __builtin_shufflevector(lo, hi, 0, 1, 2, 3, 4, 5, 6, 7,
                                         8, 9, 10, 11, 12, 13, 14, 15);
}

// ---------------------------------------------------------------------------
// bf16 GEMM strip kernel:
//   C[M, n_start + strip*NT*16 ...] (f32, row-major, ldc)
//     = A[M,K] (bf16 row-major, lda) * BT[N,K] (bf16 N-major/K-contig, ldb)
//       [+ bias[n]]
// One wave computes a 16 x (NT*16) strip with NT accumulators (A reuse xNT).
// M%16==0, K%32==0. Fragment layouts per CDNA5 ISA 7.12.2 (wave32):
//   A 16x32 bf16 : lane(l<16) row=l, elems 0-7 -> K0..7,  8-15 -> K16..23
//                  lane(l>=16) same row, K8..15 / K24..31
//   B 32x16 bf16 : lane col = l&15, same K order as A
//   C/D 16x16 f32: vgpr v, lane l -> row v + 8*(l>>4), col l&15
// ---------------------------------------------------------------------------
template <int NT>
__global__ void gemm_bf16_wmma(const __bf16* __restrict__ A,
                               const __bf16* __restrict__ BT,
                               float* __restrict__ C,
                               const float* __restrict__ bias,
                               int M, int K, int lda, int ldb, int ldc,
                               int n_start, int nstrips) {
  const int lane  = threadIdx.x & 31;
  const int gwave = blockIdx.x * (blockDim.x >> 5) + (threadIdx.x >> 5);
  if (gwave >= (M >> 4) * nstrips) return;   // wave-uniform exit

  const int m0 = (gwave / nstrips) << 4;
  const int n0 = n_start + (gwave % nstrips) * (NT * 16);

  const int arow  = m0 + (lane & 15);
  const int abase = (lane >> 4) << 3;        // 0 or 8
  const int bbase = (lane >> 4) << 3;        // 0 or 8
  const int bcol  = lane & 15;

  const __bf16* pa = A + (size_t)arow * lda + abase;
  const __bf16* pb[NT];
#pragma unroll
  for (int j = 0; j < NT; ++j)
    pb[j] = BT + (size_t)(n0 + j * 16 + bcol) * ldb + bbase;

  f32x8 acc[NT];
#pragma unroll
  for (int j = 0; j < NT; ++j) acc[j] = (f32x8){0.f,0.f,0.f,0.f,0.f,0.f,0.f,0.f};

  for (int k0 = 0; k0 < K; k0 += 32) {
    bf16x16 afrag = combine16(gload8(pa), gload8(pa + 16));
#pragma unroll
    for (int j = 0; j < NT; ++j) {
      bf16x16 bfrag = combine16(gload8(pb[j]), gload8(pb[j] + 16));
      acc[j] = __builtin_amdgcn_wmma_f32_16x16x32_bf16(
          false, afrag, false, bfrag, (short)0, acc[j], false, false);
      pb[j] += 32;
    }
    pa += 32;
  }

  const int crow0 = m0 + ((lane >> 4) << 3);
#pragma unroll
  for (int j = 0; j < NT; ++j) {
    const int cc = n0 + j * 16 + bcol;
    const float bv = bias ? gloadf(bias + cc) : 0.0f;
#pragma unroll
    for (int v = 0; v < 8; ++v)
      gstoref(C + (size_t)(crow0 + v) * ldc + cc, acc[j][v] + bv);
  }
}

// ---------------------------------------------------------------------------
// Weight prep: split W_u/W_r/W_c [D=1024, H] into state half (k<512) and
// x half (k>=512); store transposed bf16, column n = gate*512 + h.
// ---------------------------------------------------------------------------
__global__ void build_gate_weights(const float* __restrict__ Wu,
                                   const float* __restrict__ Wr,
                                   const float* __restrict__ Wc,
                                   __bf16* __restrict__ WsT,   // [1536,512]
                                   __bf16* __restrict__ WxT) { // [1536,512]
  int idx = blockIdx.x * blockDim.x + threadIdx.x;
  if (idx >= 3 * H_ * 1024) return;
  int g   = idx / (H_ * 1024);
  int rem = idx % (H_ * 1024);
  int h   = rem / 1024;
  int k   = rem % 1024;
  const float* W = (g == 0) ? Wu : (g == 1) ? Wr : Wc;
  __bf16 w = (__bf16)W[(size_t)k * H_ + h];
  int n = g * H_ + h;
  if (k < KD_) WsT[(size_t)n * KD_ + k] = w;
  else         WxT[(size_t)n * KD_ + (k - KD_)] = w;
}

__global__ void build_wout(const float* __restrict__ Wout,  // [H, V]
                           __bf16* __restrict__ WoT) {      // [V, H]
  int idx = blockIdx.x * blockDim.x + threadIdx.x;
  if (idx >= V_ * H_) return;
  int n = idx / H_;
  int k = idx % H_;
  WoT[(size_t)n * H_ + k] = (__bf16)Wout[(size_t)k * V_ + n];
}

// X[b*T+t][0:256]=embedding[token], [256:512]=baseimgfeat[b]  (bf16)
__global__ void build_X(const int* __restrict__ xTokens,
                        const float* __restrict__ emb,
                        const float* __restrict__ img,
                        __bf16* __restrict__ X) {
  int idx = blockIdx.x * blockDim.x + threadIdx.x;
  if (idx >= BT_ * KD_) return;
  int row = idx / KD_;
  int j   = idx % KD_;
  int b   = row / T_;
  float v;
  if (j < E_) v = emb[(size_t)xTokens[row] * E_ + j];
  else        v = img[(size_t)b * F_ + (j - E_)];
  X[idx] = (__bf16)v;
}

__global__ void init_state(const float* __restrict__ h0,
                           float* __restrict__ st, __bf16* __restrict__ stb) {
  int idx = blockIdx.x * blockDim.x + threadIdx.x;
  if (idx >= B_ * H_) return;
  float v = h0[idx];
  st[idx]  = v;
  stb[idx] = (__bf16)v;
}

// ---------------------------------------------------------------------------
// Per-step GRU pointwise:
//   u  = sigmoid(SPu + XPu + b_u)
//   rg = sigmoid(SPr + XPr + b_r)
//   r  = tanh(rg * (SPc + XPc) + b_c)      (bias outside gate product)
//   s' = r*(1-u) + u*s
// ---------------------------------------------------------------------------
__global__ void gru_pointwise(const float* __restrict__ SP,   // [B,1536]
                              const float* __restrict__ XP,   // [BT,1536]
                              const float* __restrict__ bu,
                              const float* __restrict__ br,
                              const float* __restrict__ bc,
                              float* __restrict__ state,      // [B,H] f32
                              __bf16* __restrict__ state_bf,  // [B,H]
                              __bf16* __restrict__ S_all,     // [BT,H]
                              float* __restrict__ final_out,  // d_out tail
                              int t) {
  int idx = blockIdx.x * blockDim.x + threadIdx.x;
  if (idx >= B_ * H_) return;
  int b = idx / H_;
  int h = idx % H_;
  size_t srow = (size_t)b * G3_;
  size_t xrow = (size_t)(b * T_ + t) * G3_;
  float su = SP[srow + h]            + XP[xrow + h];
  float sr = SP[srow + H_ + h]       + XP[xrow + H_ + h];
  float sc = SP[srow + 2 * H_ + h]   + XP[xrow + 2 * H_ + h];
  float u  = 1.0f / (1.0f + expf(-(su + bu[h])));
  float rg = 1.0f / (1.0f + expf(-(sr + br[h])));
  float r  = tanhf(rg * sc + bc[h]);
  float ns = r * (1.0f - u) + u * state[idx];
  state[idx]    = ns;
  state_bf[idx] = (__bf16)ns;
  S_all[(size_t)(b * T_ + t) * H_ + h] = (__bf16)ns;
  if (t == T_ - 1) final_out[idx] = ns;
}

// ---------------------------------------------------------------------------
extern "C" void kernel_launch(void* const* d_in, const int* in_sizes, int n_in,
                              void* d_out, int out_size, void* d_ws, size_t ws_size,
                              hipStream_t stream) {
  const int*   xTokens = (const int*)  d_in[0];
  const float* img     = (const float*)d_in[1];
  const float* h0      = (const float*)d_in[2];
  const float* emb     = (const float*)d_in[3];
  const float* Wu      = (const float*)d_in[4];
  const float* bu      = (const float*)d_in[5];
  const float* Wr      = (const float*)d_in[6];
  const float* br      = (const float*)d_in[7];
  const float* Wc      = (const float*)d_in[8];
  const float* bc      = (const float*)d_in[9];
  const float* Wout    = (const float*)d_in[10];
  const float* bout    = (const float*)d_in[11];
  float* out = (float*)d_out;
  float* final_state_out = out + (size_t)B_ * T_ * V_;

  // ---- workspace layout ----
  char* w = (char*)d_ws;
  size_t off = 0;
  auto take = [&](size_t bytes) -> char* {
    char* p = w + off;
    off = (off + bytes + 255) & ~(size_t)255;
    return p;
  };
  __bf16* WsT    = (__bf16*)take((size_t)G3_ * KD_ * 2);   // 1.5 MB
  __bf16* WxT    = (__bf16*)take((size_t)G3_ * KD_ * 2);   // 1.5 MB
  __bf16* WoT    = (__bf16*)take((size_t)V_  * H_  * 2);   // 10  MB
  __bf16* X      = (__bf16*)take((size_t)BT_ * KD_ * 2);   // 4   MB
  float*  XP     = (float*) take((size_t)BT_ * G3_ * 4);   // 24  MB
  float*  SP     = (float*) take((size_t)B_  * G3_ * 4);   // 384 KB
  float*  state  = (float*) take((size_t)B_  * H_  * 4);
  __bf16* statebf= (__bf16*)take((size_t)B_  * H_  * 2);
  __bf16* S_all  = (__bf16*)take((size_t)BT_ * H_  * 2);   // 4 MB

  const int TPB = 256;

  // GEMM launcher: full 64-wide strips with NT=4, ragged tail (16/32/48)
  // as a separate constant-NT launch so no WMMA sits under a lane predicate.
  auto launch_gemm = [&](const __bf16* A, const __bf16* Bt, float* C,
                         const float* bias, int M, int N, int K,
                         int lda, int ldb, int ldc) {
    int full = N >> 6;                       // strips of 64
    if (full > 0) {
      int waves  = (M >> 4) * full;
      int blocks = (waves * 32 + TPB - 1) / TPB;
      gemm_bf16_wmma<4><<<blocks, TPB, 0, stream>>>(
          A, Bt, C, bias, M, K, lda, ldb, ldc, 0, full);
    }
    int remt = (N & 63) >> 4;                // leftover 16-col tiles
    if (remt > 0) {
      int waves  = (M >> 4);
      int blocks = (waves * 32 + TPB - 1) / TPB;
      int nst    = full << 6;
      if (remt == 1)
        gemm_bf16_wmma<1><<<blocks, TPB, 0, stream>>>(
            A, Bt, C, bias, M, K, lda, ldb, ldc, nst, 1);
      else if (remt == 2)
        gemm_bf16_wmma<2><<<blocks, TPB, 0, stream>>>(
            A, Bt, C, bias, M, K, lda, ldb, ldc, nst, 1);
      else
        gemm_bf16_wmma<3><<<blocks, TPB, 0, stream>>>(
            A, Bt, C, bias, M, K, lda, ldb, ldc, nst, 1);
    }
  };

  // ---- prep ----
  build_gate_weights<<<(3 * H_ * 1024 + TPB - 1) / TPB, TPB, 0, stream>>>(
      Wu, Wr, Wc, WsT, WxT);
  build_wout<<<(V_ * H_ + TPB - 1) / TPB, TPB, 0, stream>>>(Wout, WoT);
  build_X<<<(BT_ * KD_ + TPB - 1) / TPB, TPB, 0, stream>>>(xTokens, emb, img, X);
  init_state<<<(B_ * H_ + TPB - 1) / TPB, TPB, 0, stream>>>(h0, state, statebf);

  // ---- time-invariant x-projection for all steps/gates (parallel GEMM) ----
  launch_gemm(X, WxT, XP, nullptr, BT_, G3_, KD_, KD_, KD_, G3_);

  // ---- sequential recurrence ----
  for (int t = 0; t < T_; ++t) {
    launch_gemm(statebf, WsT, SP, nullptr, B_, G3_, KD_, KD_, KD_, G3_);
    gru_pointwise<<<(B_ * H_ + TPB - 1) / TPB, TPB, 0, stream>>>(
        SP, XP, bu, br, bc, state, statebf, S_all, final_state_out, t);
  }

  // ---- batched logits: [4096,512] x [512,10000] + b_out -> d_out ----
  launch_gemm(S_all, WoT, out, bout, BT_, V_, KD_, KD_, KD_, V_);
}
